// BiMambaPlus_78039555769091
// MI455X (gfx1250) — matrix operand order repaired and checked by reading, not compiled, not run on hardware
//
#include <hip/hip_runtime.h>
#include <hip/hip_bf16.h>
#include <cstdint>

// ---------------------------------------------------------------------------
// BiMamba+ forward for MI455X (gfx1250, wave32, WMMA).
// Dense projections: v_wmma_f32_16x16x32_f16, f16 A + pre-converted f16 W.
// B tiles staged block-wide into LDS via global_load_async_to_lds_b128
// (ASYNCcnt, double-buffered); all ds_load_b128 fragments for a K-step are
// issued before the 8 WMMAs so DScnt waits are partial, not full.
// Input pointer order assumed = jax pytree flatten (sorted dict keys):
//   params{in_b,in_g,layers[3]{bwd{...},ffn_W1,ffn_W2,ffn_b1,ffn_b2,fwd{...},
//          n1_b,n1_g,n2_b,n2_g},norm_b,norm_g,o_W1,o_W2,o_b1,o_b2,pp_W,pp_b}, x
// ---------------------------------------------------------------------------

#define DMv   256
#define DIv   512
#define DSTv  16
#define DTRv  32
#define SEQv  6156
#define Bv    8
#define Pv    512
#define FFv   1024
#define ROWS  (Bv * Pv)      // 4096
#define EPSv  1e-5f

typedef __attribute__((ext_vector_type(16))) _Float16 v16h;
typedef __attribute__((ext_vector_type(8)))  _Float16 v8h;
typedef __attribute__((ext_vector_type(8)))  float    v8f;

union V16 { v16h v; v8h h[2]; };

__device__ __forceinline__ float sigmoidf_(float x){ return 1.f/(1.f+__expf(-x)); }
__device__ __forceinline__ float siluf_(float x){ return x * sigmoidf_(x); }
__device__ __forceinline__ float geluf_(float x){ return 0.5f*x*(1.f+erff(x*0.70710678118654752f)); }
__device__ __forceinline__ float softplusf_(float x){ return (x > 20.f) ? x : log1pf(__expf(x)); }

template<int ACT> __device__ __forceinline__ float act_(float v){
  if (ACT == 1) return siluf_(v);
  if (ACT == 2) return sigmoidf_(v);
  if (ACT == 3) return softplusf_(v);
  if (ACT == 4) return geluf_(v);
  return v;
}

// f32 -> f16 weight conversion (runs once per launch; weights stay in L2)
__global__ __launch_bounds__(256)
void cvt_kernel(const float* __restrict__ s, _Float16* __restrict__ d, int n)
{
  int i = (blockIdx.x * 256 + threadIdx.x) * 4;
  if (i >= n) return;
  float4 v = *(const float4*)(s + i);
  d[i]   = (_Float16)v.x; d[i+1] = (_Float16)v.y;
  d[i+2] = (_Float16)v.z; d[i+3] = (_Float16)v.w;
}

// ---------------------------------------------------------------------------
// WMMA GEMM:  C[M,N] = act(A[M,K](f16) @ W[N,K]^T(f16) + bias)
// 256 threads = 8 waves; wave -> 16 x (NT*16) strip; block tile 128 x (NT*16).
// B tile (NT*16 rows x 32 K halves) DMA'd block-wide into LDS with
// global_load_async_to_lds_b128, double-buffered across K-steps. All LDS
// fragment loads for a K-step are issued before the WMMAs (partial DScnt
// waits; WMMA issue hides remaining LDS latency).
// ---------------------------------------------------------------------------
template<int ACT, int NT>
__global__ __launch_bounds__(256)
void gemm_kernel(const _Float16* __restrict__ A, int lda,
                 const _Float16* __restrict__ W, const float* __restrict__ bias,
                 float* __restrict__ Cf, _Float16* __restrict__ Ch,
                 int M, int N, int K)
{
  constexpr int RB = NT * 16;               // B rows per block tile
  __shared__ _Float16 smB[2][RB][40];       // 80B row pitch

  const int tid   = threadIdx.x;
  const int lane  = tid & 31;
  const int wave  = tid >> 5;
  const int lrow  = lane & 15;
  const int lhalf = lane >> 4;
  const int tm = blockIdx.x * 128 + wave * 16;
  const int tn = blockIdx.y * RB;

  const uint32_t ldsbase = (uint32_t)(uintptr_t)(&smB[0][0][0]);

  // async-DMA one 32-wide K slice of the B tile into smB[buf]
  auto stage = [&](int k0, int buf) {
#pragma unroll
    for (int it = 0; it < NT / 4; ++it) {
      int id  = tid + it * 256;
      int seg = id & 3;                     // 16B segment within a 64B row
      int r   = id >> 2;                    // row within tile
      const _Float16* ga = W + (size_t)(tn + r) * K + k0 + seg * 8;
      uint32_t la = ldsbase + (uint32_t)(buf * RB * 80 + r * 80 + seg * 16);
      asm volatile("global_load_async_to_lds_b128 %0, %1, off"
                   :: "v"(la), "v"(ga) : "memory");
    }
  };

  v8f c[NT] = {};
  const _Float16* arow = A + (size_t)(tm + lrow) * lda;

  stage(0, 0);
  const int steps = K >> 5;
  for (int s = 0; s < steps; ++s) {
    const int k0 = s << 5;
    asm volatile("s_wait_asynccnt 0x0" ::: "memory");   // this wave's DMA landed
    __syncthreads();                                    // whole tile visible
    if (s + 1 < steps) stage(k0 + 32, (s + 1) & 1);     // overlap next DMA

    V16 a;
    // 16-bit A layout: lane<16 holds K{0..7,16..23}, lane>=16 holds K{8..15,24..31}
    a.h[0] = *(const v8h*)(arow + k0 + lhalf * 8);
    a.h[1] = *(const v8h*)(arow + k0 + 16 + lhalf * 8);
    if (k0 + 32 < K) __builtin_prefetch(arow + k0 + 32, 0, 1);

    const int buf = s & 1;
    // issue all LDS fragment loads first ...
    V16 bb[NT];
#pragma unroll
    for (int t = 0; t < NT; ++t) {
      // 16-bit B layout: lane<16 holds K{0..7,16..23}, lane>=16 holds K{8..15,24..31}
      bb[t].h[0] = *(const v8h*)(&smB[buf][t * 16 + lrow][lhalf * 8]);
      bb[t].h[1] = *(const v8h*)(&smB[buf][t * 16 + lrow][16 + lhalf * 8]);
    }
    // ... then run the WMMAs (partial DScnt waits, loads still in flight)
#pragma unroll
    for (int t = 0; t < NT; ++t)
      c[t] = __builtin_amdgcn_wmma_f32_16x16x32_f16(false, a.v, false, bb[t].v,
                                                    (short)0, c[t], false, false);
  }

  const int rbase = tm + (lhalf << 3);
#pragma unroll
  for (int t = 0; t < NT; ++t) {
    int col = tn + t * 16 + lrow;
    float bv = bias ? bias[col] : 0.f;
#pragma unroll
    for (int r = 0; r < 8; ++r) {
      float v = act_<ACT>(c[t][r] + bv);
      size_t o = (size_t)(rbase + r) * N + col;
      if (Cf) Cf[o] = v;
      if (Ch) Ch[o] = (_Float16)v;
    }
  }
}

// ------------------------- elementwise / reductions ------------------------

__global__ __launch_bounds__(256)
void meanvar_kernel(const float* __restrict__ x, float* __restrict__ mv)
{
  __shared__ float s1[256], s2[256];
  int b = blockIdx.x, t = threadIdx.x;
  float a = 0.f, a2 = 0.f;
  for (int i = t; i < SEQv; i += 256) { float v = x[(size_t)b * SEQv + i]; a += v; a2 += v * v; }
  s1[t] = a; s2[t] = a2; __syncthreads();
  for (int off = 128; off; off >>= 1) {
    if (t < off) { s1[t] += s1[t + off]; s2[t] += s2[t + off]; }
    __syncthreads();
  }
  if (t == 0) {
    float mu = s1[0] / SEQv;
    float var = s2[0] / SEQv - mu * mu;
    mv[b * 2] = mu; mv[b * 2 + 1] = rsqrtf(var + EPSv);
  }
}

__global__ __launch_bounds__(256)
void patch_kernel(const float* __restrict__ x, const float* __restrict__ mv,
                  const float* __restrict__ gp, const float* __restrict__ bp,
                  const float* __restrict__ ppW, const float* __restrict__ ppb,
                  float* __restrict__ tok, _Float16* __restrict__ tokh)
{
  int idx = blockIdx.x * 256 + threadIdx.x;                 // B*P*DM
  int d = idx & 255, p = (idx >> 8) & 511, b = idx >> 17;
  float mu = mv[b * 2], inv = mv[b * 2 + 1];
  float g = gp[0], be = bp[0];
  const float* xb = x + (size_t)b * SEQv + p * 12;
  float acc = ppb[d];
#pragma unroll
  for (int j = 0; j < 24; ++j)
    acc += ((xb[j] - mu) * inv * g + be) * ppW[d * 24 + j];
  tok[idx] = acc; tokh[idx] = (_Float16)acc;
}

// LayerNorm over C (256/512): one wave per row; optional residual adds;
// optional f32 and f16 outputs.
__global__ __launch_bounds__(256)
void ln_kernel(const float* __restrict__ x, const float* __restrict__ a1,
               const float* __restrict__ a2, const float* __restrict__ g,
               const float* __restrict__ b, float* __restrict__ of,
               _Float16* __restrict__ oh, int rows, int C)
{
  int wid  = (blockIdx.x * blockDim.x + threadIdx.x) >> 5;
  int lane = threadIdx.x & 31;
  if (wid >= rows) return;
  size_t base = (size_t)wid * C;
  int per = C >> 5;
  float vals[16];
  float s = 0.f, s2 = 0.f;
  for (int i = 0; i < per; ++i) {
    int c = lane + (i << 5);
    float v = x[base + c];
    if (a1) v += a1[base + c];
    if (a2) v += a2[base + c];
    vals[i] = v; s += v; s2 += v * v;
  }
  for (int off = 16; off; off >>= 1) { s += __shfl_xor(s, off, 32); s2 += __shfl_xor(s2, off, 32); }
  float mu = s / C;
  float inv = rsqrtf(s2 / C - mu * mu + EPSv);
  for (int i = 0; i < per; ++i) {
    int c = lane + (i << 5);
    float v = (vals[i] - mu) * inv * g[c] + b[c];
    if (of) of[base + c] = v;
    if (oh) oh[base + c] = (_Float16)v;
  }
}

// block conv: 256ch -> 512ch, kernel 3, pad(1,1), groups=256 (ci = co/2), +silu
__global__ __launch_bounds__(256)
void conv3_silu_kernel(const float* __restrict__ x, const float* __restrict__ w,
                       const float* __restrict__ cb, float* __restrict__ out)
{
  int idx = blockIdx.x * 256 + threadIdx.x;                 // B*P*DI
  int c = idx & 511, s = (idx >> 9) & 511, bb = idx >> 18;
  int ci = c >> 1;
  const float* xb = x + ((size_t)bb * Pv) * DMv + ci;
  float acc = cb[c];
#pragma unroll
  for (int k = 0; k < 3; ++k) {
    int ss = s - 1 + k;
    if (ss >= 0 && ss < Pv) acc += xb[(size_t)ss * DMv] * w[c * 3 + k];
  }
  out[idx] = siluf_(acc);
}

// mamba causal depthwise conv (DC=4, lpad=3) on xi slice of xi_z, +silu
__global__ __launch_bounds__(256)
void mconv_kernel(const float* __restrict__ xiz, const float* __restrict__ w,
                  const float* __restrict__ cb, float* __restrict__ xs,
                  _Float16* __restrict__ xsh)
{
  int idx = blockIdx.x * 256 + threadIdx.x;                 // B*P*DI
  int c = idx & 511, s = (idx >> 9) & 511, bb = idx >> 18;
  const float* xb = xiz + ((size_t)bb * Pv) * (2 * DIv) + c;
  float acc = cb[c];
#pragma unroll
  for (int k = 0; k < 4; ++k) {
    int ss = s - 3 + k;
    if (ss >= 0) acc += xb[(size_t)ss * (2 * DIv)] * w[c * 4 + k];
  }
  float v = siluf_(acc);
  xs[idx] = v; xsh[idx] = (_Float16)v;
}

// selective scan: one thread per (b,d); 16-state recurrence; Bm/Cm staged in LDS
#define SCAN_CH 64
__global__ __launch_bounds__(256)
void scan_kernel(const float* __restrict__ u, const float* __restrict__ dt,
                 const float* __restrict__ dbl, const float* __restrict__ A_log,
                 const float* __restrict__ Dp, float* __restrict__ y)
{
  int tid = threadIdx.x;
  int gid = blockIdx.x * 256 + tid;                         // 0..4095
  int b = gid >> 9, d = gid & 511;
  int bblk = blockIdx.x >> 1;                               // same b for whole block
  __shared__ float sBC[SCAN_CH][32];                        // [s][0..15]=Bm, [16..31]=Cm
  float A[DSTv];
#pragma unroll
  for (int n = 0; n < DSTv; ++n) A[n] = -__expf(A_log[d * DSTv + n]);
  float Dd = Dp[d];
  float h[DSTv] = {};
  for (int s0 = 0; s0 < Pv; s0 += SCAN_CH) {
    for (int t = tid; t < SCAN_CH * 32; t += 256) {
      int ss = t >> 5, n = t & 31;
      sBC[ss][n] = dbl[((size_t)(bblk * Pv + s0 + ss)) * 64 + 32 + n];
    }
    __syncthreads();
    for (int ss = 0; ss < SCAN_CH; ++ss) {
      size_t row = ((size_t)(b * Pv + s0 + ss)) * DIv + d;
      float dtv = dt[row], uv = u[row];
      float du = dtv * uv, acc = 0.f;
#pragma unroll
      for (int n = 0; n < DSTv; ++n) {
        h[n] = __expf(dtv * A[n]) * h[n] + du * sBC[ss][n];
        acc += h[n] * sBC[ss][16 + n];
      }
      y[row] = acc + uv * Dd;
    }
    __syncthreads();
  }
}

__global__ __launch_bounds__(256)
void ymul_kernel(const float* __restrict__ y, const float* __restrict__ xiz,
                 _Float16* __restrict__ ygh)
{
  int idx = blockIdx.x * 256 + threadIdx.x;                 // B*P*DI
  int c = idx & 511; size_t row = (size_t)(idx >> 9);
  float z = xiz[row * (2 * DIv) + DIv + c];
  ygh[idx] = (_Float16)(y[idx] * siluf_(z));
}

__global__ __launch_bounds__(256)
void combine_kernel(const float* __restrict__ sout, const float* __restrict__ xc,
                    const float* __restrict__ g, _Float16* __restrict__ combh)
{
  int idx = blockIdx.x * 256 + threadIdx.x;                 // B*P*DI
  int c = idx & 511; size_t row = (size_t)(idx >> 9);
  float gv = g[row * DMv + (c >> 1)];                       // gv = sigmoid(...)
  float v = sout[idx] * (gv * sigmoidf_(gv)) + xc[idx] * (1.f - gv);
  combh[idx] = (_Float16)v;
}

__global__ __launch_bounds__(256)
void flip_kernel(const float* __restrict__ in, float* __restrict__ of,
                 _Float16* __restrict__ oh)
{
  int idx = blockIdx.x * 256 + threadIdx.x;                 // B*P*DM
  int c = idx & 255, s = (idx >> 8) & 511, bb = idx >> 17;
  float v = in[((size_t)bb * Pv + (Pv - 1 - s)) * DMv + c];
  if (of) of[idx] = v;
  if (oh) oh[idx] = (_Float16)v;
}

__global__ __launch_bounds__(256)
void pool_kernel(const float* __restrict__ t, float* __restrict__ pooled)
{
  int b = blockIdx.x, d = threadIdx.x;
  float s = 0.f;
  for (int ss = 0; ss < Pv; ++ss) s += t[((size_t)b * Pv + ss) * DMv + d];
  pooled[b * DMv + d] = s * (1.f / Pv);
}

__global__ __launch_bounds__(128)
void head_kernel(const float* __restrict__ pooled, const float* __restrict__ W1,
                 const float* __restrict__ b1, const float* __restrict__ W2,
                 const float* __restrict__ b2, float* __restrict__ out)
{
  int b = blockIdx.x, t = threadIdx.x;                      // 128 threads
  __shared__ float sh[128];
  float acc = b1[t];
  for (int j = 0; j < DMv; ++j) acc += pooled[b * DMv + j] * W1[t * DMv + j];
  sh[t] = geluf_(acc) * W2[t];
  __syncthreads();
  for (int off = 64; off; off >>= 1) { if (t < off) sh[t] += sh[t + off]; __syncthreads(); }
  if (t == 0) out[b] = sh[0] + b2[0];
}

// ------------------------------ host orchestration -------------------------

struct MambaP { const float *A_log,*D,*conv_b,*conv_w,*dt_W,*dt_b,*in_W,*out_W,*x_W; };
struct BlkP   { const float *conv_b,*conv_w,*gate_W,*gate_b,*ln1_b,*ln1_g,*ln2_b,*ln2_g;
                MambaP m; const float *op_W,*op_b; };
struct EncP   { BlkP bwd; const float *ffn_W1,*ffn_W2,*ffn_b1,*ffn_b2; BlkP fwd;
                const float *n1_b,*n1_g,*n2_b,*n2_g; };
// f16 copies of GEMM weights
struct WBlk   { _Float16 *in_W,*x_W,*dt_W,*out_W,*gate_W,*op_W; };
struct WEnc   { WBlk bwd, fwd; _Float16 *ffn_W1,*ffn_W2; };

extern "C" void kernel_launch(void* const* d_in, const int* in_sizes, int n_in,
                              void* d_out, int out_size, void* d_ws, size_t ws_size,
                              hipStream_t stream)
{
  (void)in_sizes; (void)out_size;
  if (n_in < 149) return;

  int ii = 0;
  auto F = [&]() { return (const float*)d_in[ii++]; };
  auto getM = [&](MambaP& m) {
    m.A_log=F(); m.D=F(); m.conv_b=F(); m.conv_w=F(); m.dt_W=F(); m.dt_b=F();
    m.in_W=F(); m.out_W=F(); m.x_W=F();
  };
  auto getB = [&](BlkP& bl) {
    bl.conv_b=F(); bl.conv_w=F(); bl.gate_W=F(); bl.gate_b=F();
    bl.ln1_b=F(); bl.ln1_g=F(); bl.ln2_b=F(); bl.ln2_g=F();
    getM(bl.m); bl.op_W=F(); bl.op_b=F();
  };

  const float* in_b = F();
  const float* in_g = F();
  EncP L[3];
  for (int l = 0; l < 3; ++l) {
    getB(L[l].bwd);
    L[l].ffn_W1=F(); L[l].ffn_W2=F(); L[l].ffn_b1=F(); L[l].ffn_b2=F();
    getB(L[l].fwd);
    L[l].n1_b=F(); L[l].n1_g=F(); L[l].n2_b=F(); L[l].n2_g=F();
  }
  const float* norm_b = F(); const float* norm_g = F();
  const float* o_W1 = F(); const float* o_W2 = F();
  const float* o_b1 = F(); const float* o_b2 = F();
  const float* pp_W = F(); const float* pp_b = F();
  const float* x = F();

  // ---- workspace bump allocator ----
  uint8_t* wsb = (uint8_t*)d_ws; size_t off = 0;
  auto allocB = [&](size_t bytes) -> void* {
    off = (off + 255) & ~(size_t)255;
    void* p = wsb + off; off += bytes; return p;
  };
  auto allocF = [&](size_t n) { return (float*)allocB(n * 4); };
  auto allocH = [&](size_t n) { return (_Float16*)allocB(n * 2); };

  float*    mv     = allocF(16);
  float*    tok    = allocF((size_t)ROWS * DMv);
  _Float16* tokh   = allocH((size_t)ROWS * DMv);
  float*    xflip  = allocF((size_t)ROWS * DMv);
  _Float16* xfliph = allocH((size_t)ROWS * DMv);
  float*    xcpre  = allocF((size_t)ROWS * DIv);
  float*    xc     = allocF((size_t)ROWS * DIv);
  _Float16* xch    = allocH((size_t)ROWS * DIv);
  float*    xiz    = allocF((size_t)ROWS * 2 * DIv);
  float*    xis    = allocF((size_t)ROWS * DIv);
  _Float16* xish   = allocH((size_t)ROWS * DIv);
  float*    dbl    = allocF((size_t)ROWS * 64);
  _Float16* dblh   = allocH((size_t)ROWS * 64);
  float*    dtb    = allocF((size_t)ROWS * DIv);
  float*    yb     = allocF((size_t)ROWS * DIv);
  _Float16* ygh    = allocH((size_t)ROWS * DIv);
  _Float16* combh  = allocH((size_t)ROWS * DIv);
  float*    fo     = allocF((size_t)ROWS * DMv);
  float*    bo     = allocF((size_t)ROWS * DMv);
  float*    xr     = allocF((size_t)ROWS * DMv);
  _Float16* xrh    = allocH((size_t)ROWS * DMv);
  _Float16* ffhh   = allocH((size_t)ROWS * FFv);
  float*    ffn    = allocF((size_t)ROWS * DMv);
  float*    pooled = allocF(Bv * DMv);
  // safe aliases (lifetimes disjoint, stream-ordered)
  float* sout  = xis;    // xis dead after scan
  float* gbuf  = yb;     // y dead after ymul
  float* opout = dtb;    // dt dead after scan
  float* bflip = xcpre;  // xcpre dead after ln2; reused post-blocks

  // f16 weight copies (converted once per launch; L2-resident thereafter)
  auto cvt = [&](const float* src, size_t n) -> _Float16* {
    _Float16* d = allocH(n);
    cvt_kernel<<<(unsigned)((n / 4 + 255) / 256), 256, 0, stream>>>(src, d, (int)n);
    return d;
  };
  auto cvtBlk = [&](const BlkP& P, WBlk& W) {
    W.in_W   = cvt(P.m.in_W,  (size_t)2 * DIv * DIv);
    W.x_W    = cvt(P.m.x_W,   (size_t)64 * DIv);
    W.dt_W   = cvt(P.m.dt_W,  (size_t)DIv * DTRv);
    W.out_W  = cvt(P.m.out_W, (size_t)DIv * DIv);
    W.gate_W = cvt(P.gate_W,  (size_t)DMv * DMv);
    W.op_W   = cvt(P.op_W,    (size_t)DMv * DIv);
  };
  WEnc WL[3];
  for (int l = 0; l < 3; ++l) {
    cvtBlk(L[l].fwd, WL[l].fwd);
    cvtBlk(L[l].bwd, WL[l].bwd);
    WL[l].ffn_W1 = cvt(L[l].ffn_W1, (size_t)FFv * DMv);
    WL[l].ffn_W2 = cvt(L[l].ffn_W2, (size_t)DMv * FFv);
  }
  if (off > ws_size) return;

  const dim3 blk(256);
  auto run_block = [&](const BlkP& P, const WBlk& W, const float* xin,
                       const _Float16* xinh, float* outp) {
    conv3_silu_kernel<<<ROWS * DIv / 256, blk, 0, stream>>>(xin, P.conv_w, P.conv_b, xcpre);
    ln_kernel<<<ROWS / 8, blk, 0, stream>>>(xcpre, nullptr, nullptr, P.ln2_g, P.ln2_b,
                                            xc, xch, ROWS, DIv);
    gemm_kernel<0,8><<<dim3(ROWS/128, (2*DIv)/128), blk, 0, stream>>>(
        xch, DIv, W.in_W, nullptr, xiz, nullptr, ROWS, 2*DIv, DIv);
    mconv_kernel<<<ROWS * DIv / 256, blk, 0, stream>>>(xiz, P.m.conv_w, P.m.conv_b, xis, xish);
    gemm_kernel<0,4><<<dim3(ROWS/128, 1), blk, 0, stream>>>(
        xish, DIv, W.x_W, nullptr, dbl, dblh, ROWS, 64, DIv);
    gemm_kernel<3,8><<<dim3(ROWS/128, DIv/128), blk, 0, stream>>>(
        dblh, 64, W.dt_W, P.m.dt_b, dtb, nullptr, ROWS, DIv, DTRv);
    scan_kernel<<<ROWS / 256, blk, 0, stream>>>(xis, dtb, dbl, P.m.A_log, P.m.D, yb);
    ymul_kernel<<<ROWS * DIv / 256, blk, 0, stream>>>(yb, xiz, ygh);
    gemm_kernel<0,8><<<dim3(ROWS/128, DIv/128), blk, 0, stream>>>(
        ygh, DIv, W.out_W, nullptr, sout, nullptr, ROWS, DIv, DIv);
    gemm_kernel<2,8><<<dim3(ROWS/128, DMv/128), blk, 0, stream>>>(
        xinh, DMv, W.gate_W, P.gate_b, gbuf, nullptr, ROWS, DMv, DMv);
    combine_kernel<<<ROWS * DIv / 256, blk, 0, stream>>>(sout, xc, gbuf, combh);
    gemm_kernel<0,8><<<dim3(ROWS/128, DMv/128), blk, 0, stream>>>(
        combh, DIv, W.op_W, P.op_b, opout, nullptr, ROWS, DMv, DIv);
    ln_kernel<<<ROWS / 8, blk, 0, stream>>>(opout, xin, nullptr, P.ln1_g, P.ln1_b,
                                            outp, nullptr, ROWS, DMv);
  };

  // ---- stem: instance-norm + patchify + patch projection ----
  meanvar_kernel<<<Bv, blk, 0, stream>>>(x, mv);
  patch_kernel<<<ROWS * DMv / 256, blk, 0, stream>>>(x, mv, in_g, in_b, pp_W, pp_b, tok, tokh);

  // ---- encoder layers ----
  for (int l = 0; l < 3; ++l) {
    run_block(L[l].fwd, WL[l].fwd, tok, tokh, fo);
    flip_kernel<<<ROWS * DMv / 256, blk, 0, stream>>>(tok, xflip, xfliph);
    run_block(L[l].bwd, WL[l].bwd, xflip, xfliph, bo);
    flip_kernel<<<ROWS * DMv / 256, blk, 0, stream>>>(bo, bflip, nullptr);
    ln_kernel<<<ROWS / 8, blk, 0, stream>>>(bflip, fo, tok, L[l].n1_g, L[l].n1_b,
                                            xr, xrh, ROWS, DMv);
    gemm_kernel<4,8><<<dim3(ROWS/128, FFv/128), blk, 0, stream>>>(
        xrh, DMv, WL[l].ffn_W1, L[l].ffn_b1, nullptr, ffhh, ROWS, FFv, DMv);
    gemm_kernel<0,8><<<dim3(ROWS/128, DMv/128), blk, 0, stream>>>(
        ffhh, FFv, WL[l].ffn_W2, L[l].ffn_b2, ffn, nullptr, ROWS, DMv, FFv);
    ln_kernel<<<ROWS / 8, blk, 0, stream>>>(ffn, xr, nullptr, L[l].n2_g, L[l].n2_b,
                                            tok, tokh, ROWS, DMv);
  }

  // ---- head ----
  ln_kernel<<<ROWS / 8, blk, 0, stream>>>(tok, nullptr, nullptr, norm_g, norm_b,
                                          xr, nullptr, ROWS, DMv);
  pool_kernel<<<Bv, blk, 0, stream>>>(xr, pooled);
  head_kernel<<<Bv, 128, 0, stream>>>(pooled, o_W1, o_b1, o_W2, o_b2, (float*)d_out);
}
